// LPModel_85263690760360
// MI455X (gfx1250) — matrix operand order repaired and checked by reading, not compiled, not run on hardware
//
#include <hip/hip_runtime.h>
#include <hip/hip_bf16.h>

typedef __attribute__((ext_vector_type(16))) _Float16 v16h;
typedef __attribute__((ext_vector_type(8)))  _Float16 v8h;
typedef __attribute__((ext_vector_type(8)))  float    v8f;

#define N_NODES 30000
#define DIM     128
#define T_LINKS 3000
#define NQ      6000        // 3000 left rows + 3000 right rows
#define KNEG    75
#define EPSF    1e-7f
#define MAXSQ   50.0f
#define NTILE   1875        // 30000 / 16
#define NWAVES  8           // 256 threads / wave32
#define FIXS    16777216.0f // 2^24 fixed-point scale for deterministic accumulation

// ---------------------------------------------------------------------------
// Split f32 embeddings into (hi, lo) f16 pair: x = hi + lo  (Markidis split)
// ---------------------------------------------------------------------------
__global__ void split_embeddings(const float* __restrict__ emb,
                                 _Float16* __restrict__ Eh,
                                 _Float16* __restrict__ El) {
  const int n = N_NODES * DIM;
  for (int i = blockIdx.x * blockDim.x + threadIdx.x; i < n;
       i += gridDim.x * blockDim.x) {
    float x = emb[i];
    _Float16 h = (_Float16)x;
    Eh[i] = h;
    El[i] = (_Float16)(x - (float)h);
  }
}

// ---------------------------------------------------------------------------
// Build query rows (component 0 negated -> Minkowski dot == plain dot),
// split to f16 hi/lo, and compute D[i] = sqdist(left,right) + GAMMA (exact f32)
// ---------------------------------------------------------------------------
__global__ void build_queries(const float* __restrict__ emb,
                              const float* __restrict__ cptr,
                              const int*   __restrict__ links,
                              _Float16* __restrict__ Qh,
                              _Float16* __restrict__ Ql,
                              float* __restrict__ Dq) {
  __shared__ float red[128];
  const int q = blockIdx.x;   // 0..5999
  const int t = threadIdx.x;  // 0..127
  const int li  = (q < T_LINKS) ? q : (q - T_LINKS);
  const int src = (q < T_LINKS) ? links[2 * li] : links[2 * li + 1];

  float x  = emb[(size_t)src * DIM + t];
  float xs = (t == 0) ? -x : x;                    // fold Minkowski sign into query
  _Float16 h = (_Float16)xs;
  Qh[(size_t)q * DIM + t] = h;
  Ql[(size_t)q * DIM + t] = (_Float16)(xs - (float)h);

  if (q < T_LINKS) {
    const int l = links[2 * q], r = links[2 * q + 1];
    float a = emb[(size_t)l * DIM + t];
    float b = emb[(size_t)r * DIM + t];
    float term = a * b;
    if (t == 0) term = -term;                      // Minkowski dot
    red[t] = term;
    __syncthreads();
    for (int s = 64; s > 0; s >>= 1) {
      if (t < s) red[t] += red[t + s];
      __syncthreads();
    }
    if (t == 0) {
      float cs    = cptr[0];
      float theta = fmaxf(-red[0] * cs, 1.0f + EPSF);
      float ac    = acoshf(theta);
      float sq    = fminf((1.0f / cs) * ac * ac, MAXSQ);
      float Dv    = sq + 1.0f;                     // + GAMMA
      Dq[q] = Dv;
      Dq[q + T_LINKS] = Dv;                        // right half uses same D
    }
  }
}

// ---------------------------------------------------------------------------
// 16x16 dot-product tile via split-f16 WMMA (3 x v_wmma_f32_16x16x32_f16 / K=32)
// ---------------------------------------------------------------------------
__device__ __forceinline__ v8f tile_prod(const _Float16* __restrict__ ehp,
                                         const _Float16* __restrict__ elp,
                                         const _Float16 (*Qhs)[DIM],
                                         const _Float16 (*Qls)[DIM],
                                         int rm, int half) {
  v8f c = {};
#pragma unroll
  for (int kb = 0; kb < DIM; kb += 32) {
    // A fragment (queries): lane holds row rm; e 0..7 -> K=kb+8h+e, 8..15 -> K=kb+16+8h+e
    v8h a0h = *(const v8h*)&Qhs[rm][kb + 8 * half];
    v8h a1h = *(const v8h*)&Qhs[rm][kb + 16 + 8 * half];
    v8h a0l = *(const v8h*)&Qls[rm][kb + 8 * half];
    v8h a1l = *(const v8h*)&Qls[rm][kb + 16 + 8 * half];
    v16h Ah, Al;
#pragma unroll
    for (int e = 0; e < 8; ++e) {
      Ah[e] = a0h[e]; Ah[e + 8] = a1h[e];
      Al[e] = a0l[e]; Al[e + 8] = a1l[e];
    }
    // B fragment (embeddings as columns): lane holds col rm; e 0..7 -> K=kb+8h+e, 8..15 -> K=kb+16+8h+e
    v8h b0h = *(const v8h*)(ehp + kb + 8 * half);
    v8h b1h = *(const v8h*)(ehp + kb + 16 + 8 * half);
    v8h b0l = *(const v8h*)(elp + kb + 8 * half);
    v8h b1l = *(const v8h*)(elp + kb + 16 + 8 * half);
    v16h Bh, Bl;
#pragma unroll
    for (int e = 0; e < 8; ++e) {
      Bh[e] = b0h[e]; Bh[e + 8] = b1h[e];
      Bl[e] = b0l[e]; Bl[e + 8] = b1l[e];
    }

    c = __builtin_amdgcn_wmma_f32_16x16x32_f16(false, Ah, false, Bh, (short)0, c, false, false);
    c = __builtin_amdgcn_wmma_f32_16x16x32_f16(false, Ah, false, Bl, (short)0, c, false, false);
    c = __builtin_amdgcn_wmma_f32_16x16x32_f16(false, Al, false, Bh, (short)0, c, false, false);
  }
  return c;
}

// descending-in-prod sortable key (smallest key == largest dot == smallest dist)
__device__ __forceinline__ unsigned prod_key(float prod) {
  unsigned s = __float_as_uint(prod);
  return ~(s ^ ((unsigned)(((int)s) >> 31) | 0x80000000u));
}

// ---------------------------------------------------------------------------
// Main: per 16 query rows, radix-select 75 smallest distances over 30000 cols
// 4 histogram passes on key bytes + 1 accumulation pass. All WMMA GEMM sweeps.
// ---------------------------------------------------------------------------
__global__ __launch_bounds__(256) void topk_loss(
    const _Float16* __restrict__ Eh, const _Float16* __restrict__ El,
    const _Float16* __restrict__ Qh, const _Float16* __restrict__ Ql,
    const float* __restrict__ Dq, const float* __restrict__ cptr,
    float* __restrict__ rowacc) {
  __shared__ _Float16 Qhs[16][DIM];
  __shared__ _Float16 Qls[16][DIM];
  __shared__ unsigned cnt[16][256];
  __shared__ unsigned pref[16];
  __shared__ int rankr[16];
  __shared__ float Dsh[16];
  __shared__ unsigned long long acc64[16];

  const int tid  = threadIdx.x;
  const int lane = tid & 31;
  const int wid  = tid >> 5;
  const int half = lane >> 4;
  const int rm   = lane & 15;
  const int wg   = blockIdx.x;
  const float cs = cptr[0];
  const float Kc = 1.0f / cs;

  for (int i = tid; i < 16 * DIM; i += 256) {
    int r = i >> 7, k = i & (DIM - 1);
    Qhs[r][k] = Qh[(size_t)(wg * 16 + r) * DIM + k];
    Qls[r][k] = Ql[(size_t)(wg * 16 + r) * DIM + k];
  }
  if (tid < 16) {
    Dsh[tid]   = Dq[wg * 16 + tid];
    pref[tid]  = 0u;
    rankr[tid] = KNEG;
    acc64[tid] = 0ull;
  }
  __syncthreads();

  // ---- 4 radix passes: find exact 32-bit key of the 75th-smallest distance
  for (int pass = 0; pass < 4; ++pass) {
    for (int i = tid; i < 16 * 256; i += 256) ((unsigned*)cnt)[i] = 0u;
    __syncthreads();
    const int bshift = 24 - 8 * pass;
    const int pshift = 32 - 8 * pass;   // pass 0 guarded below

    // cache this sweep's row prefixes in registers (stable between barriers)
    unsigned prefReg[8];
#pragma unroll
    for (int v = 0; v < 8; ++v) prefReg[v] = pref[v + 8 * half];

    for (int tile = wid; tile < NTILE; tile += NWAVES) {
      const int col = tile * 16 + rm;
      v8f p = tile_prod(Eh + (size_t)col * DIM, El + (size_t)col * DIM,
                        Qhs, Qls, rm, half);
#pragma unroll
      for (int v = 0; v < 8; ++v) {
        unsigned key = prod_key(p[v]);
        int row = v + 8 * half;
        bool take = (pass == 0) ? true : ((key >> pshift) == prefReg[v]);
        if (take) atomicAdd(&cnt[row][(key >> bshift) & 255], 1u);
      }
    }
    __syncthreads();
    if (tid < 16) {  // per-row 256-bin scan: pick bin containing rank r
      int r = rankr[tid];
      unsigned cum = 0, chosen = 255;
      for (int b = 0; b < 256; ++b) {
        unsigned cb = cnt[tid][b];
        if (cum + cb >= (unsigned)r) { chosen = (unsigned)b; break; }
        cum += cb;
      }
      rankr[tid] = r - (int)cum;                 // how many ties at threshold to take
      pref[tid]  = (pref[tid] << 8) | chosen;
    }
    __syncthreads();
  }

  // ---- accumulation pass: only strictly-below-threshold values hit acoshf
  {
    unsigned tReg[8];
    float    dReg[8];
#pragma unroll
    for (int v = 0; v < 8; ++v) {
      tReg[v] = pref[v + 8 * half];
      dReg[v] = Dsh[v + 8 * half];
    }
    for (int tile = wid; tile < NTILE; tile += NWAVES) {
      const int col = tile * 16 + rm;
      v8f p = tile_prod(Eh + (size_t)col * DIM, El + (size_t)col * DIM,
                        Qhs, Qls, rm, half);
#pragma unroll
      for (int v = 0; v < 8; ++v) {
        float prod   = p[v];
        unsigned key = prod_key(prod);
        if (key < tReg[v]) {                     // among the 75 smallest (non-tie)
          float theta = fmaxf(-prod * cs, 1.0f + EPSF);
          float ac    = acoshf(theta);
          float sq    = fminf(Kc * ac * ac, MAXSQ);
          float fx    = dReg[v] - sq;
          if (fx > 0.0f)                         // relu
            atomicAdd(&acc64[v + 8 * half], (unsigned long long)(fx * FIXS));
        }
      }
    }
  }
  __syncthreads();

  if (tid < 16) {  // ties at the exact threshold key: all identical values
    unsigned T    = pref[tid];
    unsigned uasc = ~T;
    unsigned s    = (uasc & 0x80000000u) ? (uasc ^ 0x80000000u) : ~uasc;
    float prodT   = __uint_as_float(s);
    float theta   = fmaxf(-prodT * cs, 1.0f + EPSF);
    float ac      = acoshf(theta);
    float sqT     = fminf(Kc * ac * ac, MAXSQ);
    float fx      = fmaxf(Dsh[tid] - sqT, 0.0f);
    unsigned long long tot =
        acc64[tid] + (unsigned long long)rankr[tid] * (unsigned long long)(fx * FIXS);
    rowacc[wg * 16 + tid] = (float)((double)tot * (1.0 / 16777216.0));
  }
}

// ---------------------------------------------------------------------------
// Deterministic final reduction of the 6000 per-row sums
// ---------------------------------------------------------------------------
__global__ void reduce_out(const float* __restrict__ rowacc,
                           float* __restrict__ out) {
  __shared__ float red[256];
  float s = 0.0f;
  for (int i = threadIdx.x; i < NQ; i += 256) s += rowacc[i];
  red[threadIdx.x] = s;
  __syncthreads();
  for (int st = 128; st > 0; st >>= 1) {
    if (threadIdx.x < st) red[threadIdx.x] += red[threadIdx.x + st];
    __syncthreads();
  }
  if (threadIdx.x == 0) out[0] = red[0] / (2.0f * KNEG * T_LINKS);
}

// ---------------------------------------------------------------------------
extern "C" void kernel_launch(void* const* d_in, const int* in_sizes, int n_in,
                              void* d_out, int out_size, void* d_ws, size_t ws_size,
                              hipStream_t stream) {
  (void)in_sizes; (void)n_in; (void)out_size; (void)ws_size;
  const float* emb  = (const float*)d_in[0];
  const float* cptr = (const float*)d_in[1];
  const int*   lnk  = (const int*)d_in[2];

  char* ws = (char*)d_ws;
  size_t off = 0;
  _Float16* Eh = (_Float16*)(ws + off); off += (size_t)N_NODES * DIM * 2;  // 7.68 MB
  _Float16* El = (_Float16*)(ws + off); off += (size_t)N_NODES * DIM * 2;  // 7.68 MB
  _Float16* Qh = (_Float16*)(ws + off); off += (size_t)NQ * DIM * 2;       // 1.54 MB
  _Float16* Ql = (_Float16*)(ws + off); off += (size_t)NQ * DIM * 2;       // 1.54 MB
  float* Dq     = (float*)(ws + off);   off += (size_t)NQ * 4;
  float* rowacc = (float*)(ws + off);   off += (size_t)NQ * 4;

  split_embeddings<<<512, 256, 0, stream>>>(emb, Eh, El);
  build_queries<<<NQ, 128, 0, stream>>>(emb, cptr, lnk, Qh, Ql, Dq);
  topk_loss<<<NQ / 16, 256, 0, stream>>>(Eh, El, Qh, Ql, Dq, cptr, rowacc);
  reduce_out<<<1, 256, 0, stream>>>(rowacc, (float*)d_out);
}
